// BertHAttention1D_9792525435288
// MI455X (gfx1250) — matrix operand order repaired and checked by reading, not compiled, not run on hardware
//
#include <hip/hip_runtime.h>
#include <hip/hip_bf16.h>
#include <float.h>

// ---------------- problem constants ----------------
#define NBATCH 4
#define SEQ    4096          // power of two -> no padding path
#define HIDDIM 1024
#define NH     16
#define DH     64
#define BH     (NBATCH * NH) // 64 head-batches
#define BSZ    16
#define NLEV   7             // lens: 4096,2048,...,64
#define TOTTOK 8128          // sum of all level lengths
#define QSCALE 0.125f        // DH^-0.5

typedef __bf16 bf16_t;
typedef __attribute__((ext_vector_type(16))) __bf16     v16bf;
typedef __attribute__((ext_vector_type(8)))  float      v8f;
typedef __attribute__((ext_vector_type(4)))  unsigned   v4u;
typedef __attribute__((ext_vector_type(8)))  int        v8i;
typedef __attribute__((ext_vector_type(4)))  int        v4i;

__device__ __host__ inline int tokoff(int l) { return 2 * SEQ - ((2 * SEQ) >> l); }

// --- WMMA bf16 16x16x32 per-lane element -> K index maps (ISA 7.12.2) ---
__device__ inline int amap_k(int lane, int e) {
  int kb = (lane >> 4) * 8;
  return (e < 8) ? (kb + e) : (kb + 16 + (e - 8));
}
__device__ inline int bmap_k(int lane, int e) { return amap_k(lane, e); }

// ---------------- Tensor Data Mover: 2D bf16 tile -> LDS ----------------
// Packs a D# per CDNA5 ISA and issues
// tensor_load_to_lds. Tracked by TENSORcnt.  Tile written row-major,
// tile_d0 contiguous elements per row, tile_d1 rows, stride0 elements
// between tensor rows.  global_addr points at the TILE start.
#if __has_builtin(__builtin_amdgcn_tensor_load_to_lds)
#define HAVE_TDM 1
__device__ inline void tdm_load_2d_bf16(unsigned lds_off, const bf16_t* tile,
                                        unsigned tensor_d0, unsigned tensor_d1,
                                        unsigned tile_d0, unsigned tile_d1,
                                        unsigned long long stride0_elems) {
  unsigned long long ga = (unsigned long long)(uintptr_t)tile;
  v4u g0;
  g0[0] = 1u;                                   // count=1, user descriptor
  g0[1] = lds_off;                              // LDS byte offset
  g0[2] = (unsigned)(ga & 0xffffffffull);       // global_addr[31:0]
  g0[3] = (unsigned)((ga >> 32) & 0x01ffffffull) | (2u << 30);  // addr[56:32]|type=2
  v8i g1;
  g1[0] = (int)(1u << 16);                      // workgroup_mask=0, data_size=1 (2B)
  g1[1] = (int)((tensor_d0 & 0xffffu) << 16);   // [63:48]=tensor_dim0[15:0]
  g1[2] = (int)((tensor_d0 >> 16) | ((tensor_d1 & 0xffffu) << 16));
  g1[3] = (int)((tensor_d1 >> 16) | (tile_d0 << 16));          // [127:112]=tile_dim0
  g1[4] = (int)(tile_d1 & 0xffffu);             // [143:128]=tile_dim1, tile_dim2=0
  g1[5] = (int)(stride0_elems & 0xffffffffull); // tensor_dim0_stride[31:0]
  g1[6] = (int)((stride0_elems >> 32) & 0xffffull);  // stride[47:32], dim1_stride=0
  g1[7] = 0;
  v4i z4 = {0, 0, 0, 0};
#if __clang_major__ >= 23
  v8i z8 = {0, 0, 0, 0, 0, 0, 0, 0};            // amdgpu-toolchain 6-arg form
  __builtin_amdgcn_tensor_load_to_lds(g0, g1, z4, z4, z8, 0);
#else
  __builtin_amdgcn_tensor_load_to_lds(g0, g1, z4, z4, 0);      // ROCm 7.2 5-arg form
#endif
}
#else
#define HAVE_TDM 0
#endif

// ---------------- fp32 -> bf16 cast pre-pass ----------------
__global__ void cast_bf16_kernel(const float* __restrict__ src,
                                 bf16_t* __restrict__ dst, size_t n4) {
  size_t i = (size_t)blockIdx.x * blockDim.x + threadIdx.x;
  if (i >= n4) return;
  float4 v = *(const float4*)(src + i * 4);
  dst[i * 4 + 0] = (bf16_t)v.x;
  dst[i * 4 + 1] = (bf16_t)v.y;
  dst[i * 4 + 2] = (bf16_t)v.z;
  dst[i * 4 + 3] = (bf16_t)v.w;
}

// ---------------- QKV projection GEMM (bf16 WMMA, TDM-staged) ----------------
// Out[m][n] = (sum_k X[m][k]*W[n][k] + bias[n]) * scale, written bf16 into the
// head-split level-0 [bh][TOTTOK][DH] layout.
#define BM 128
#define BN 64
#define BK 32
#define NIT (HIDDIM / BK)

__global__ __launch_bounds__(256) void qkv_gemm_kernel(
    const bf16_t* __restrict__ Xb, const bf16_t* __restrict__ Wb,
    const float* __restrict__ bias, float scale, bf16_t* __restrict__ Out) {
  __shared__ bf16_t As[2][BM][BK];   // 2 x 8KB
  __shared__ bf16_t Bs[2][BN][BK];   // 2 x 4KB

  const int tid  = threadIdx.x;
  const int lane = tid & 31;
  const int wave = tid >> 5;
  const int wm = wave & 3;   // 4 waves along M
  const int wn = wave >> 2;  // 2 waves along N
  const int m0 = blockIdx.x * BM;
  const int n0 = blockIdx.y * BN;

  v8f acc[2][2] = {};

#if HAVE_TDM
  // prologue: DMA tile pair 0 into buffer 0 (single issue from wave 0)
  if (wave == 0) {
    tdm_load_2d_bf16((unsigned)(size_t)&As[0][0][0], Xb + (size_t)m0 * HIDDIM,
                     HIDDIM, NBATCH * SEQ, BK, BM, HIDDIM);
    tdm_load_2d_bf16((unsigned)(size_t)&Bs[0][0][0], Wb + (size_t)n0 * HIDDIM,
                     HIDDIM, HIDDIM, BK, BN, HIDDIM);
  }
#endif

  for (int k = 0; k < NIT; ++k) {
    const int buf = k & 1;
#if HAVE_TDM
    if (wave == 0) {
      if (k + 1 < NIT) {  // prefetch next tile pair into the other buffer
        const int kn = (k + 1) * BK;
        tdm_load_2d_bf16((unsigned)(size_t)&As[buf ^ 1][0][0],
                         Xb + (size_t)m0 * HIDDIM + kn, HIDDIM, NBATCH * SEQ, BK,
                         BM, HIDDIM);
        tdm_load_2d_bf16((unsigned)(size_t)&Bs[buf ^ 1][0][0],
                         Wb + (size_t)n0 * HIDDIM + kn, HIDDIM, HIDDIM, BK, BN,
                         HIDDIM);
        __builtin_amdgcn_s_wait_tensorcnt(2);  // current pair done, next in flight
      } else {
        __builtin_amdgcn_s_wait_tensorcnt(0);
      }
    }
    __syncthreads();
#else
    // fallback staging: plain bf16 copies (8 per thread A, 4 per thread B)
    __syncthreads();
    for (int i = 0; i < 2; ++i) {
      int idx = tid + i * 256;                // 512 x 8-bf16 chunks for A
      int row = idx >> 2, c8 = idx & 3;
      *(ulonglong1*)&As[buf][row][c8 * 8] =
          *(const ulonglong1*)(Xb + (size_t)(m0 + row) * HIDDIM + k * BK + c8 * 8);
    }
    {
      int row = tid >> 2, c8 = tid & 3;       // 256 x 8-bf16 chunks for B
      *(ulonglong1*)&Bs[buf][row][c8 * 8] =
          *(const ulonglong1*)(Wb + (size_t)(n0 + row) * HIDDIM + k * BK + c8 * 8);
    }
    __syncthreads();
#endif

    v16bf afrag[2], bfrag[2];
    for (int tm = 0; tm < 2; ++tm) {
      int row = wm * 32 + tm * 16 + (lane & 15);
      for (int e = 0; e < 16; ++e) afrag[tm][e] = As[buf][row][amap_k(lane, e)];
    }
    for (int tn = 0; tn < 2; ++tn) {
      int col = wn * 32 + tn * 16 + (lane & 15);
      for (int e = 0; e < 16; ++e) bfrag[tn][e] = Bs[buf][col][bmap_k(lane, e)];
    }
    for (int tm = 0; tm < 2; ++tm)
      for (int tn = 0; tn < 2; ++tn)
        acc[tm][tn] = __builtin_amdgcn_wmma_f32_16x16x32_bf16(
            false, afrag[tm], false, bfrag[tn], (short)0, acc[tm][tn], false, false);
    __syncthreads();
  }

  // epilogue: bias + scale, scatter into head-split level-0 layout
  for (int tm = 0; tm < 2; ++tm)
    for (int tn = 0; tn < 2; ++tn)
      for (int i = 0; i < 8; ++i) {
        int mrow = m0 + wm * 32 + tm * 16 + i + ((lane >> 4) * 8);
        int ncol = n0 + wn * 32 + tn * 16 + (lane & 15);
        float v = (acc[tm][tn][i] + bias[ncol]) * scale;
        int b = mrow >> 12, t = mrow & (SEQ - 1);
        int h = ncol >> 6, d = ncol & (DH - 1);
        Out[((size_t)(b * NH + h) * TOTTOK + t) * DH + d] = (bf16_t)v;
      }
}

// ---------------- init: zero Y/A accumulators, build level-0 mask ----------------
__global__ void init_kernel(float* __restrict__ Y, float* __restrict__ Ad,
                            unsigned char* __restrict__ M,
                            const float* __restrict__ amask) {
  size_t idx = (size_t)blockIdx.x * blockDim.x + threadIdx.x;
  size_t ytot = (size_t)BH * SEQ * DH;
  if (idx < ytot) Y[idx] = 0.0f;
  if (idx < (size_t)BH * SEQ) {
    Ad[idx] = 0.0f;
    int bh = (int)(idx / SEQ), t = (int)(idx % SEQ);
    int b = bh >> 4;
    M[(size_t)bh * TOTTOK + t] = (amask[(size_t)b * SEQ + t] >= 0.0f) ? 1 : 0;
  }
}

// ---------------- hierarchical coarsening: level l -> l+1 ----------------
__global__ void coarsen_kernel(bf16_t* __restrict__ Q, bf16_t* __restrict__ K,
                               bf16_t* __restrict__ V, unsigned char* __restrict__ M,
                               int l) {
  int lenn = SEQ >> (l + 1);
  size_t idx = (size_t)blockIdx.x * blockDim.x + threadIdx.x;
  size_t total = (size_t)BH * lenn * DH;
  if (idx >= total) return;
  int d = (int)(idx % DH);
  int i = (int)((idx / DH) % lenn);
  int bh = (int)(idx / ((size_t)DH * lenn));
  int offs = tokoff(l), offd = tokoff(l + 1);

  size_t mrow = (size_t)bh * TOTTOK;
  float m0 = (float)M[mrow + offs + 2 * i];
  float m1 = (float)M[mrow + offs + 2 * i + 1];
  float cnt = m0 + m1;
  float inv = (cnt == 0.0f) ? 0.0f : (1.0f / fmaxf(cnt, 1.0f));

  size_t s0 = (mrow + offs + 2 * i) * DH + d;
  size_t s1 = s0 + DH;
  size_t dd = (mrow + offd + i) * DH + d;
  Q[dd] = (bf16_t)(((float)Q[s0] * m0 + (float)Q[s1] * m1) * inv);
  K[dd] = (bf16_t)(((float)K[s0] * m0 + (float)K[s1] * m1) * inv);
  V[dd] = (bf16_t)(((float)V[s0] * m0 + (float)V[s1] * m1));  // masked SUM
  if (d == 0) M[mrow + offd + i] = (unsigned char)((m0 + m1) > 0.0f ? 1 : 0);
}

// ---------------- per-level 16x16 block attention (1 wave per block) ----------------
__global__ __launch_bounds__(32) void block_attn_kernel(
    const bf16_t* __restrict__ Q, const bf16_t* __restrict__ K,
    const bf16_t* __restrict__ V, const unsigned char* __restrict__ M,
    float* __restrict__ Y, float* __restrict__ Ad, int l) {
  const int lane = threadIdx.x;
  const int len = SEQ >> l;
  const int nb = len / BSZ;
  const int blk = blockIdx.x % nb;
  const int bh  = blockIdx.x / nb;
  const int off = tokoff(l);
  const int kb  = (l == 0) ? blk : (blk ^ 1);  // sibling block for coarse levels

  const size_t mrow  = (size_t)bh * TOTTOK;
  const size_t qbase = (mrow + off + blk * 16) * DH;
  const size_t kbase = (mrow + off + kb * 16) * DH;
  const unsigned char* mq = M + mrow + off + blk * 16;
  const unsigned char* mk = M + mrow + off + kb * 16;

  __shared__ float  Slds[16][17];
  __shared__ bf16_t Alds[16][16];

  // ---- S = q (16x64) @ k^T (64x16): two K=32 WMMAs ----
  const int r = lane & 15;
  v8f s = {};
  for (int half = 0; half < 2; ++half) {
    v16bf qf, kf;
    for (int e = 0; e < 16; ++e) {
      qf[e] = Q[qbase + (size_t)r * DH + half * 32 + amap_k(lane, e)];
      kf[e] = K[kbase + (size_t)r * DH + half * 32 + bmap_k(lane, e)];
    }
    s = __builtin_amdgcn_wmma_f32_16x16x32_bf16(false, qf, false, kf, (short)0, s,
                                                false, false);
  }

  // ---- mask + spill S (C layout: M = vgpr + 8*(lane>=16), N = lane&15) ----
  for (int i = 0; i < 8; ++i) {
    int row = i + ((lane >> 4) * 8);
    int col = lane & 15;
    Slds[row][col] = (mq[row] && mk[col]) ? s[i] : -FLT_MAX;
  }
  __syncthreads();

  // ---- rowwise exp(S - max); denominator with 2^l replication ----
  if (lane < 16) {
    float mx = -FLT_MAX;
    for (int j = 0; j < 16; ++j) mx = fmaxf(mx, Slds[lane][j]);
    float rowsum = 0.0f;
    for (int j = 0; j < 16; ++j) {
      float a = __expf(Slds[lane][j] - mx);  // fully-masked row -> exp(0)=1 (as ref)
      Alds[lane][j] = (bf16_t)a;
      rowsum += a;
    }
    int rep = 1 << l;
    size_t t0 = (size_t)bh * SEQ + ((size_t)(blk * 16 + lane) << l);
    for (int rr = 0; rr < rep; ++rr) Ad[t0 + rr] += rowsum;
  }
  __syncthreads();

  // ---- y = A (16x16) @ v (16x64): K padded 16->32, four WMMAs ----
  v16bf af;
  for (int e = 0; e < 16; ++e) {
    int kk = amap_k(lane, e);
    af[e] = (kk < 16) ? Alds[r][kk] : (bf16_t)0.0f;
  }
  const int rep = 1 << l;
  for (int c = 0; c < 4; ++c) {
    v16bf vf;
    for (int e = 0; e < 16; ++e) {
      int kk = bmap_k(lane, e);  // key index j
      vf[e] = (kk < 16) ? V[kbase + (size_t)kk * DH + c * 16 + (lane & 15)]
                        : (bf16_t)0.0f;
    }
    v8f y = {};
    y = __builtin_amdgcn_wmma_f32_16x16x32_bf16(false, af, false, vf, (short)0, y,
                                                false, false);
    for (int i = 0; i < 8; ++i) {
      int row = i + ((lane >> 4) * 8);
      int col = c * 16 + (lane & 15);
      size_t tfine = ((size_t)(blk * 16 + row) << l);
      float yy = y[i];
      for (int rr = 0; rr < rep; ++rr)
        Y[((size_t)bh * SEQ + tfine + rr) * DH + col] += yy;  // unique owner
    }
  }
}

// ---------------- finalize: out = Y / (A + eps), merge heads ----------------
__global__ void finalize_kernel(const float* __restrict__ Y,
                                const float* __restrict__ Ad,
                                float* __restrict__ out) {
  size_t idx = (size_t)blockIdx.x * blockDim.x + threadIdx.x;
  size_t total = (size_t)BH * SEQ * DH;
  if (idx >= total) return;
  int d = (int)(idx % DH);
  int t = (int)((idx / DH) % SEQ);
  int bh = (int)(idx / ((size_t)DH * SEQ));
  int b = bh >> 4, h = bh & 15;
  float denom = Ad[(size_t)bh * SEQ + t] + 1e-9f;
  out[((size_t)b * SEQ + t) * HIDDIM + h * DH + d] = Y[idx] / denom;
}

// ---------------- host launcher ----------------
extern "C" void kernel_launch(void* const* d_in, const int* in_sizes, int n_in,
                              void* d_out, int out_size, void* d_ws, size_t ws_size,
                              hipStream_t stream) {
  const float* X     = (const float*)d_in[0];
  const float* amask = (const float*)d_in[1];
  const float* Wq    = (const float*)d_in[2];
  const float* bq    = (const float*)d_in[3];
  const float* Wk    = (const float*)d_in[4];
  const float* bk    = (const float*)d_in[5];
  const float* Wv    = (const float*)d_in[6];
  const float* bv    = (const float*)d_in[7];
  float* out = (float*)d_out;

  // workspace carve-up (~307 MB)
  char* ws = (char*)d_ws;
  size_t o = 0;
  auto carve = [&](size_t bytes) -> void* {
    void* p = ws + o;
    o = (o + bytes + 255) & ~(size_t)255;
    return p;
  };
  const size_t lvl_elems = (size_t)BH * TOTTOK * DH;
  bf16_t* Qlv = (bf16_t*)carve(lvl_elems * sizeof(bf16_t));
  bf16_t* Klv = (bf16_t*)carve(lvl_elems * sizeof(bf16_t));
  bf16_t* Vlv = (bf16_t*)carve(lvl_elems * sizeof(bf16_t));
  unsigned char* Mlv = (unsigned char*)carve((size_t)BH * TOTTOK);
  float* Yacc = (float*)carve((size_t)BH * SEQ * DH * sizeof(float));
  float* Aden = (float*)carve((size_t)BH * SEQ * sizeof(float));
  bf16_t* Xb  = (bf16_t*)carve((size_t)NBATCH * SEQ * HIDDIM * sizeof(bf16_t));
  bf16_t* Wqb = (bf16_t*)carve((size_t)HIDDIM * HIDDIM * sizeof(bf16_t));
  bf16_t* Wkb = (bf16_t*)carve((size_t)HIDDIM * HIDDIM * sizeof(bf16_t));
  bf16_t* Wvb = (bf16_t*)carve((size_t)HIDDIM * HIDDIM * sizeof(bf16_t));

  // 0) one-time fp32 -> bf16 casts (X reused by all three GEMMs)
  {
    size_t n4 = (size_t)NBATCH * SEQ * HIDDIM / 4;
    cast_bf16_kernel<<<(unsigned)((n4 + 255) / 256), 256, 0, stream>>>(X, Xb, n4);
    size_t w4 = (size_t)HIDDIM * HIDDIM / 4;
    unsigned wg = (unsigned)((w4 + 255) / 256);
    cast_bf16_kernel<<<wg, 256, 0, stream>>>(Wq, Wqb, w4);
    cast_bf16_kernel<<<wg, 256, 0, stream>>>(Wk, Wkb, w4);
    cast_bf16_kernel<<<wg, 256, 0, stream>>>(Wv, Wvb, w4);
  }

  // 1) QKV projections (TDM-staged bf16 WMMA); q gets the DH^-0.5 scale
  dim3 gg((NBATCH * SEQ) / BM, HIDDIM / BN);
  qkv_gemm_kernel<<<gg, 256, 0, stream>>>(Xb, Wqb, bq, QSCALE, Qlv);
  qkv_gemm_kernel<<<gg, 256, 0, stream>>>(Xb, Wkb, bk, 1.0f, Klv);
  qkv_gemm_kernel<<<gg, 256, 0, stream>>>(Xb, Wvb, bv, 1.0f, Vlv);

  // 2) init accumulators + level-0 mask
  {
    size_t tot = (size_t)BH * SEQ * DH;
    init_kernel<<<(unsigned)((tot + 255) / 256), 256, 0, stream>>>(Yacc, Aden, Mlv,
                                                                   amask);
  }

  // 3) build coarse levels 1..6
  for (int l = 0; l < NLEV - 1; ++l) {
    size_t tot = (size_t)BH * (SEQ >> (l + 1)) * DH;
    coarsen_kernel<<<(unsigned)((tot + 255) / 256), 256, 0, stream>>>(Qlv, Klv, Vlv,
                                                                      Mlv, l);
  }

  // 4) block attention at every level, accumulating numer/denom at finest res
  for (int l = 0; l < NLEV; ++l) {
    unsigned nblocks = (unsigned)(BH * ((SEQ >> l) / BSZ));
    block_attn_kernel<<<nblocks, 32, 0, stream>>>(Qlv, Klv, Vlv, Mlv, Yacc, Aden, l);
  }

  // 5) normalize and merge heads
  {
    size_t tot = (size_t)BH * SEQ * DH;
    finalize_kernel<<<(unsigned)((tot + 255) / 256), 256, 0, stream>>>(Yacc, Aden,
                                                                       out);
  }
}
